// searchMaskCrossAttention_41669772705963
// MI455X (gfx1250) — compile-verified, matrix operand order checked by reading, not run on hardware
//
#include <hip/hip_runtime.h>

// ---------------------------------------------------------------------------
// Problem constants (reference: searchMaskCrossAttention)
// ---------------------------------------------------------------------------
constexpr int B       = 16;
constexpr int LQ      = 1024;
constexpr int LKV     = 1424;
constexpr int LKV_PAD = 1440;           // round up to multiple of 32
constexpr int DIM     = 384;
constexpr int HEADS   = 8;
constexpr int HD      = 48;             // head dim
constexpr int TEM     = 400;
constexpr float SCALE = 0.14433756729740643f;  // 48^-0.5

typedef __bf16 bf16_t;
typedef __attribute__((ext_vector_type(16))) __bf16 bf16x16;
typedef __attribute__((ext_vector_type(8)))  float  f32x8;

__device__ __forceinline__ bf16_t bf16_zero() {
    return __builtin_bit_cast(__bf16, (unsigned short)0);
}

__device__ __forceinline__ bf16_t f2bf(float f) {
    unsigned u = __builtin_bit_cast(unsigned, f);
    unsigned r = (u + 0x7FFFu + ((u >> 16) & 1u)) >> 16;
    return __builtin_bit_cast(__bf16, (unsigned short)r);
}

// k index inside a 16-bit 16x32 A/B fragment, element i (0..15) of v16bf.
// Lanes 0-15 hold K = {0..7, 16..23}; lanes 16-31 hold K = {8..15, 24..31}.
__device__ __forceinline__ int frag_k_idx(int lane, int i) {
    int kh = (lane >> 4) & 1;
    return ((i < 8) ? 0 : 16) + kh * 8 + (i & 7);
}

__device__ __forceinline__ bf16x16 load_frag(const bf16_t* p, long rowStride, long kStride) {
    int lane = threadIdx.x & 31;
    const bf16_t* base = p + (long)(lane & 15) * rowStride;
    bf16x16 f;
#pragma unroll
    for (int i = 0; i < 16; ++i)
        f[i] = base[(long)frag_k_idx(lane, i) * kStride];
    return f;
}

// Elements with (kbase + k) >= kmax are zero (head-dim 48 padded to K=64).
__device__ __forceinline__ bf16x16 load_frag_guard(const bf16_t* p, long rowStride,
                                                   long kStride, int kbase, int kmax) {
    int lane = threadIdx.x & 31;
    const bf16_t* base = p + (long)(lane & 15) * rowStride;
    bf16x16 f;
#pragma unroll
    for (int i = 0; i < 16; ++i) {
        int k = frag_k_idx(lane, i);
        f[i] = (kbase + k < kmax) ? base[(long)k * kStride] : bf16_zero();
    }
    return f;
}

__device__ __forceinline__ f32x8 wmma_bf16(bf16x16 a, bf16x16 b, f32x8 c) {
    return __builtin_amdgcn_wmma_f32_16x16x32_bf16(
        false, a, false, b, (short)0, c, false, false);
}

// CDNA5 async copy: 16 bytes global -> LDS per lane, tracked by ASYNCcnt.
__device__ __forceinline__ void async_copy16(unsigned lds_off, const void* gptr) {
    asm volatile("global_load_async_to_lds_b128 %0, %1, off"
                 :: "v"(lds_off), "v"((unsigned long long)(uintptr_t)gptr)
                 : "memory");
}

// ---------------------------------------------------------------------------
// Stage 0: f32 -> bf16 conversions
// ---------------------------------------------------------------------------
__global__ void k_cvt(const float* __restrict__ s, bf16_t* __restrict__ d, long n) {
    long i = (long)blockIdx.x * blockDim.x + threadIdx.x;
    long stride = (long)gridDim.x * blockDim.x;
    for (; i < n; i += stride) d[i] = f2bf(s[i]);
}

// kv: [B, LKV, DIM] f32 -> [B, LKV_PAD, DIM] bf16 with zeroed pad rows
__global__ void k_cvt_kv(const float* __restrict__ s, bf16_t* __restrict__ d) {
    long n = (long)B * LKV_PAD * DIM;
    long i = (long)blockIdx.x * blockDim.x + threadIdx.x;
    long stride = (long)gridDim.x * blockDim.x;
    for (; i < n; i += stride) {
        long c  = i % DIM;
        long kv = (i / DIM) % LKV_PAD;
        long b  = i / ((long)DIM * LKV_PAD);
        d[i] = (kv < LKV) ? f2bf(s[(b * LKV + kv) * DIM + c]) : bf16_zero();
    }
}

// ---------------------------------------------------------------------------
// GEMM core: one wave computes a 16(M) x 64(N) tile, A row-major [*,DIM],
// W row-major [N,DIM] (i.e. X @ W^T).  Fully unrolled K loop (12 chunks of
// 32) so the scheduler can hoist loads ahead of WMMAs with partial
// s_wait_loadcnt and without rotate-by-copy register moves.
// ---------------------------------------------------------------------------
__device__ __forceinline__ void gemm_tile_16x64(const bf16_t* __restrict__ A,
                                                const bf16_t* __restrict__ W,
                                                int n0, f32x8 acc[4]) {
#pragma unroll
    for (int kc = 0; kc < DIM; kc += 32) {
        bf16x16 a = load_frag(A + kc, DIM, 1);
#pragma unroll
        for (int j = 0; j < 4; ++j) {
            bf16x16 bw = load_frag(W + (long)(n0 + j * 16) * DIM + kc, DIM, 1);
            acc[j] = wmma_bf16(a, bw, acc[j]);
        }
    }
}

// ---------------------------------------------------------------------------
// Stage 1: qh = (q @ q_w^T + q_b) -> bf16 head-major [B,H,LQ,48]
// ---------------------------------------------------------------------------
__global__ __launch_bounds__(128) void k_gemm_q(const bf16_t* __restrict__ qbf,
                                                const bf16_t* __restrict__ wq,
                                                const float*  __restrict__ qb,
                                                bf16_t* __restrict__ qh) {
    int b = blockIdx.x, mt = blockIdx.y, nt = blockIdx.z;
    int w = threadIdx.x >> 5, lane = threadIdx.x & 31;
    int m0 = mt * 64 + w * 16;
    f32x8 acc[4] = {};
    gemm_tile_16x64(qbf + ((long)b * LQ + m0) * DIM, wq, nt * 64, acc);

    int hi = lane >> 4, cl = lane & 15;
#pragma unroll
    for (int j = 0; j < 4; ++j) {
        int col = nt * 64 + j * 16 + cl;
        int hh = col / HD, dd = col % HD;
        float bias = qb[col];
#pragma unroll
        for (int r = 0; r < 8; ++r) {
            int row = m0 + r + 8 * hi;
            qh[(((long)b * HEADS + hh) * LQ + row) * HD + dd] = f2bf(acc[j][r] + bias);
        }
    }
}

// ---------------------------------------------------------------------------
// Stage 2: kvp = kv @ kv_w^T + kv_b ; split K / (V + masks) head-major,
// zero-fill kv pad rows.
// ---------------------------------------------------------------------------
__global__ __launch_bounds__(128) void k_gemm_kv(const bf16_t* __restrict__ kvbf,
                                                 const bf16_t* __restrict__ wkv,
                                                 const float*  __restrict__ kvb,
                                                 const float*  __restrict__ temMask,
                                                 const float*  __restrict__ searchMask,
                                                 bf16_t* __restrict__ kh,
                                                 bf16_t* __restrict__ vh) {
    int b = blockIdx.x, mt = blockIdx.y, nt = blockIdx.z;
    int w = threadIdx.x >> 5, lane = threadIdx.x & 31;
    int m0 = mt * 64 + w * 16;
    if (m0 >= LKV_PAD) return;                    // no barriers below -> safe
    f32x8 acc[4] = {};
    gemm_tile_16x64(kvbf + ((long)b * LKV_PAD + m0) * DIM, wkv, nt * 64, acc);

    int hi = lane >> 4, cl = lane & 15;
#pragma unroll
    for (int j = 0; j < 4; ++j) {
        int col = nt * 64 + j * 16 + cl;          // 0..767
        float bias = kvb[col];
        bool isK = col < DIM;
        int c = isK ? col : col - DIM;
        int hh = c / HD, dd = c % HD;
#pragma unroll
        for (int r = 0; r < 8; ++r) {
            int kv = m0 + r + 8 * hi;
            long oidx = (((long)b * HEADS + hh) * LKV_PAD + kv) * HD + dd;
            if (kv >= LKV) {
                if (isK) kh[oidx] = bf16_zero(); else vh[oidx] = bf16_zero();
            } else {
                float val = acc[j][r] + bias;
                if (isK) {
                    kh[oidx] = f2bf(val);
                } else {
                    float mk = (kv < TEM)
                        ? temMask[((long)b * TEM + kv) * DIM + c]
                        : searchMask[((long)b * (LKV - TEM) + (kv - TEM)) * DIM + c];
                    vh[oidx] = f2bf(val + mk);
                }
            }
        }
    }
}

// ---------------------------------------------------------------------------
// Stage 3: flash attention with CDNA5 async-LDS staging.
// grid (B*H, LQ/64), block 128 (4 waves, 16 q-rows each).
// Per kv-chunk of 32: the block asynchronously stages K (32x48 bf16) and
// V (32x48) into a double-buffered LDS region (global_load_async_to_lds_b128,
// 3 ops/thread, ASYNCcnt), transposes V into a bank-conflict-padded [48][36]
// tile, then: 4 WMMAs for S, online softmax (__shfl_xor row reductions),
// per-wave LDS round-trip for P, 3 WMMAs for O += P V.
// ---------------------------------------------------------------------------
constexpr int RAWK_ELEMS = 32 * HD;                 // 1536
constexpr int RAWBUF     = 2 * RAWK_ELEMS;          // K+V per buffer (3072)
constexpr int VT_STRIDE  = 36;                      // padded (avoids bank conflicts)
constexpr int VT_OFF     = 2 * RAWBUF;              // 6144
constexpr int PL_OFF     = VT_OFF + HD * VT_STRIDE; // 6144 + 1728
constexpr int SMEM_ELEMS = PL_OFF + 4 * 16 * 32;    // + 2048

__global__ __launch_bounds__(128) void k_attn(const bf16_t* __restrict__ qh,
                                              const bf16_t* __restrict__ kh,
                                              const bf16_t* __restrict__ vh,
                                              const float*  __restrict__ pos,
                                              bf16_t* __restrict__ xbf) {
    __shared__ __align__(16) bf16_t smem[SMEM_ELEMS];
    int bh = blockIdx.x;
    int b = bh >> 3, h = bh & 7;
    int tid = threadIdx.x;
    int w = tid >> 5, lane = tid & 31;
    int hi = lane >> 4, cl = lane & 15;
    int q0 = blockIdx.y * 64 + w * 16;

    const bf16_t* Qb = qh + ((long)bh * LQ + q0) * HD;
    bf16x16 aq0 = load_frag(Qb, HD, 1);                        // d = 0..31
    bf16x16 aq1 = load_frag_guard(Qb + 32, HD, 1, 32, HD);     // d = 32..47 (+pad)

    f32x8 o0 = {}, o1 = {}, o2 = {};
    float mrow[8], lrow[8];
#pragma unroll
    for (int r = 0; r < 8; ++r) { mrow[r] = -1e30f; lrow[r] = 0.f; }

    const bf16_t* kSlab = kh + (long)bh * LKV_PAD * HD;
    const bf16_t* vSlab = vh + (long)bh * LKV_PAD * HD;
    bf16_t* pl = &smem[PL_OFF + w * 16 * 32];
    constexpr int NCHUNK = (LKV + 31) / 32;                    // 45

    // stage a 32-row K+V chunk into LDS buffer `buf` (3072 B each, contiguous;
    // 384 x 16B units over 128 threads = exactly 3 async ops per thread)
    auto stage = [&](int chunk, int buf) {
        const char* kg = (const char*)(kSlab + (long)chunk * 32 * HD);
        const char* vg = (const char*)(vSlab + (long)chunk * 32 * HD);
        unsigned kOff = (unsigned)(uintptr_t)(const void*)&smem[buf * RAWBUF];
        unsigned vOff = (unsigned)(uintptr_t)(const void*)&smem[buf * RAWBUF + RAWK_ELEMS];
#pragma unroll
        for (int it = 0; it < 3; ++it) {
            int u = tid + it * 128;               // 0..383
            bool isK = u < 192;
            int uu = isK ? u : u - 192;
            async_copy16((isK ? kOff : vOff) + uu * 16,
                         (isK ? kg : vg) + uu * 16);
        }
    };

    stage(0, 0);

    for (int kc = 0; kc < NCHUNK; ++kc) {
        int kv0 = kc * 32;
        int buf = kc & 1;
        if (kc + 1 < NCHUNK) {
            stage(kc + 1, buf ^ 1);               // prefetch next chunk
            asm volatile("s_wait_asynccnt 0x3" ::: "memory");
        } else {
            asm volatile("s_wait_asynccnt 0x0" ::: "memory");
        }
        __syncthreads();                          // current buffer ready

        // cooperative V transpose: raw [32][48] -> VT [48][36-padded]
        {
            const bf16_t* vr = &smem[buf * RAWBUF + RAWK_ELEMS];
            bf16_t* vt = &smem[VT_OFF];
#pragma unroll
            for (int it = 0; it < 12; ++it) {
                int e = tid + it * 128;           // 0..1535
                int kvr = e / HD, dd = e % HD;
                vt[dd * VT_STRIDE + kvr] = vr[e];
            }
        }
        __syncthreads();

        // ---- S = Q K^T for two 16-wide kv tiles (K frags from LDS)
        const bf16_t* Kl = &smem[buf * RAWBUF];
        f32x8 s0 = {}, s1 = {};
        {
            bf16x16 bk0 = load_frag(Kl, HD, 1);
            bf16x16 bk1 = load_frag_guard(Kl + 32, HD, 1, 32, HD);
            s0 = wmma_bf16(aq0, bk0, s0);
            s0 = wmma_bf16(aq1, bk1, s0);
            bk0 = load_frag(Kl + 16 * HD, HD, 1);
            bk1 = load_frag_guard(Kl + 16 * HD + 32, HD, 1, 32, HD);
            s1 = wmma_bf16(aq0, bk0, s1);
            s1 = wmma_bf16(aq1, bk1, s1);
        }

        // ---- scale + attn_pos + mask + online softmax
        int col0 = kv0 + cl, col1 = kv0 + 16 + cl;
        bool ok0 = col0 < LKV, ok1 = col1 < LKV;
#pragma unroll
        for (int r = 0; r < 8; ++r) {
            int row = q0 + r + 8 * hi;
            const float* prow = pos + ((long)h * LQ + row) * LKV;
            float pv0 = prow[ok0 ? col0 : 0];
            float pv1 = prow[ok1 ? col1 : 0];
            float v0 = ok0 ? s0[r] * SCALE + pv0 : -1e30f;
            float v1 = ok1 ? s1[r] * SCALE + pv1 : -1e30f;
            float mx = fmaxf(v0, v1);
#pragma unroll
            for (int off = 1; off < 16; off <<= 1)
                mx = fmaxf(mx, __shfl_xor(mx, off, 32));
            float mnew  = fmaxf(mrow[r], mx);
            float alpha = __expf(mrow[r] - mnew);
            mrow[r] = mnew;
            float p0 = __expf(v0 - mnew);
            float p1 = __expf(v1 - mnew);
            float rs = p0 + p1;
#pragma unroll
            for (int off = 1; off < 16; off <<= 1)
                rs += __shfl_xor(rs, off, 32);
            lrow[r] = lrow[r] * alpha + rs;
            o0[r] *= alpha; o1[r] *= alpha; o2[r] *= alpha;
            int rr = r + 8 * hi;                  // C-layout row -> LDS row
            pl[rr * 32 + cl]      = f2bf(p0);
            pl[rr * 32 + 16 + cl] = f2bf(p1);
        }
        // wave-private LDS round-trip: C-layout P -> A-fragment layout
        asm volatile("s_wait_dscnt 0x0" ::: "memory");
        bf16x16 ap = load_frag(pl, 32, 1);

        // ---- O += P V  (V frags from transposed LDS tile, contiguous in k)
        const bf16_t* Vt = &smem[VT_OFF];
        bf16x16 bv0 = load_frag(Vt,                  VT_STRIDE, 1);
        bf16x16 bv1 = load_frag(Vt + 16 * VT_STRIDE, VT_STRIDE, 1);
        bf16x16 bv2 = load_frag(Vt + 32 * VT_STRIDE, VT_STRIDE, 1);
        o0 = wmma_bf16(ap, bv0, o0);
        o1 = wmma_bf16(ap, bv1, o1);
        o2 = wmma_bf16(ap, bv2, o2);

        __syncthreads();        // protect raw buffer + VT before next overwrite
    }

    // ---- epilogue: x[b, q, h*48 + d] = O / l
#pragma unroll
    for (int r = 0; r < 8; ++r) {
        int row = q0 + r + 8 * hi;
        float inv = 1.0f / lrow[r];
        long base = ((long)b * LQ + row) * DIM + h * HD;
        xbf[base + cl]      = f2bf(o0[r] * inv);
        xbf[base + 16 + cl] = f2bf(o1[r] * inv);
        xbf[base + 32 + cl] = f2bf(o2[r] * inv);
    }
}

// ---------------------------------------------------------------------------
// Stage 4: out = x @ proj_w^T + proj_b  (f32 output)
// ---------------------------------------------------------------------------
__global__ __launch_bounds__(128) void k_gemm_out(const bf16_t* __restrict__ xbf,
                                                  const bf16_t* __restrict__ wp,
                                                  const float*  __restrict__ pb,
                                                  float* __restrict__ out) {
    int b = blockIdx.x, mt = blockIdx.y, nt = blockIdx.z;
    int w = threadIdx.x >> 5, lane = threadIdx.x & 31;
    int m0 = mt * 64 + w * 16;
    f32x8 acc[4] = {};
    gemm_tile_16x64(xbf + ((long)b * LQ + m0) * DIM, wp, nt * 64, acc);

    int hi = lane >> 4, cl = lane & 15;
#pragma unroll
    for (int j = 0; j < 4; ++j) {
        int col = nt * 64 + j * 16 + cl;
        float bias = pb[col];
#pragma unroll
        for (int r = 0; r < 8; ++r) {
            int row = m0 + r + 8 * hi;
            out[((long)b * LQ + row) * DIM + col] = acc[j][r] + bias;
        }
    }
}

// ---------------------------------------------------------------------------
// Host side
// ---------------------------------------------------------------------------
static inline size_t align256(size_t x) { return (x + 255) & ~(size_t)255; }

extern "C" void kernel_launch(void* const* d_in, const int* in_sizes, int n_in,
                              void* d_out, int out_size, void* d_ws, size_t ws_size,
                              hipStream_t stream) {
    const float* tem_mask    = (const float*)d_in[0];
    const float* search_mask = (const float*)d_in[1];
    const float* q           = (const float*)d_in[2];
    const float* kv          = (const float*)d_in[3];
    const float* attn_pos    = (const float*)d_in[4];
    const float* q_w         = (const float*)d_in[5];
    const float* q_b         = (const float*)d_in[6];
    const float* kv_w        = (const float*)d_in[7];
    const float* kv_b        = (const float*)d_in[8];
    const float* proj_w      = (const float*)d_in[9];
    const float* proj_b      = (const float*)d_in[10];
    float* out = (float*)d_out;

    // workspace carve-up (bf16 staging buffers)
    char* p = (char*)d_ws;
    const size_t QBF  = (size_t)B * LQ * DIM * 2;
    const size_t KVBF = (size_t)B * LKV_PAD * DIM * 2;
    const size_t WQ   = (size_t)DIM * DIM * 2;
    const size_t WKV  = (size_t)2 * DIM * DIM * 2;
    const size_t QH   = (size_t)B * HEADS * LQ * HD * 2;
    const size_t KH   = (size_t)B * HEADS * LKV_PAD * HD * 2;

    bf16_t* qbf  = (bf16_t*)p; p += align256(QBF);
    bf16_t* kvbf = (bf16_t*)p; p += align256(KVBF);
    bf16_t* wq   = (bf16_t*)p; p += align256(WQ);
    bf16_t* wkv  = (bf16_t*)p; p += align256(WKV);
    bf16_t* wp   = (bf16_t*)p; p += align256(WQ);
    bf16_t* qh   = (bf16_t*)p; p += align256(QH);
    bf16_t* kh   = (bf16_t*)p; p += align256(KH);
    bf16_t* vh   = (bf16_t*)p; p += align256(KH);
    bf16_t* xbf  = (bf16_t*)p; p += align256(QBF);

    // Stage 0: conversions
    k_cvt<<<4096, 256, 0, stream>>>(q, qbf, (long)B * LQ * DIM);
    k_cvt_kv<<<4096, 256, 0, stream>>>(kv, kvbf);
    k_cvt<<<576, 256, 0, stream>>>(q_w, wq, (long)DIM * DIM);
    k_cvt<<<1152, 256, 0, stream>>>(kv_w, wkv, (long)2 * DIM * DIM);
    k_cvt<<<576, 256, 0, stream>>>(proj_w, wp, (long)DIM * DIM);

    // Stage 1: Q projection
    k_gemm_q<<<dim3(B, LQ / 64, DIM / 64), 128, 0, stream>>>(qbf, wq, q_b, qh);

    // Stage 2: KV projection + V masks
    k_gemm_kv<<<dim3(B, (LKV_PAD + 63) / 64, (2 * DIM) / 64), 128, 0, stream>>>(
        kvbf, wkv, kv_b, tem_mask, search_mask, kh, vh);

    // Stage 3: flash attention (async-LDS staged)
    k_attn<<<dim3(B * HEADS, LQ / 64), 128, 0, stream>>>(qh, kh, vh, attn_pos, xbf);

    // Stage 4: output projection
    k_gemm_out<<<dim3(B, LQ / 64, DIM / 64), 128, 0, stream>>>(xbf, wp, proj_b, out);
}
